// Attention_59356448031540
// MI455X (gfx1250) — matrix operand order repaired and checked by reading, not compiled, not run on hardware
//
#include <hip/hip_runtime.h>

// ---------------------------------------------------------------------------
// MI455X (gfx1250, wave32) fused attention:
//   bf16 WMMA GEMMs (fp32 accum) + fused RoPE + flash-style causal softmax.
//   32x64 per-wave GEMM tiles and 32-row flash q-tiles for 2x operand reuse.
// ---------------------------------------------------------------------------

typedef __bf16 v16bf __attribute__((ext_vector_type(16)));
typedef __bf16 v8bf  __attribute__((ext_vector_type(8)));
typedef __bf16 v4bf  __attribute__((ext_vector_type(4)));
typedef float  v8f   __attribute__((ext_vector_type(8)));

constexpr int B_  = 4;
constexpr int S_  = 2048;
constexpr int D_  = 1024;
constexpr int H_  = 16;
constexpr int DH_ = 64;
constexpr int M_  = B_ * S_;   // 8192 flattened (b,s) rows

__device__ __forceinline__ v16bf mk16(v8bf lo, v8bf hi) {
    union { v16bf v; v8bf h[2]; } u;
    u.h[0] = lo; u.h[1] = hi;
    return u.v;
}

__device__ __forceinline__ v8f wmma_bf16(v16bf a, v16bf b, v8f c) {
    // v_wmma_f32_16x16x32_bf16  (neg_a, A, neg_b, B, c_mod, C, reuse_a, reuse_b)
    return __builtin_amdgcn_wmma_f32_16x16x32_bf16(false, a, false, b, (short)0, c, false, false);
}

__device__ __forceinline__ float rmax16(float v) {
#pragma unroll
    for (int m = 1; m < 16; m <<= 1) v = fmaxf(v, __shfl_xor(v, m, 32));
    return v;
}
__device__ __forceinline__ float rsum16(float v) {
#pragma unroll
    for (int m = 1; m < 16; m <<= 1) v += __shfl_xor(v, m, 32);
    return v;
}

// ---------------------------------------------------------------------------
// 1) fp32 -> bf16 convert (vectorized 4-wide; all sizes are multiples of 4)
// ---------------------------------------------------------------------------
__global__ __launch_bounds__(256) void cvt_bf16_kernel(
    const float* __restrict__ src, __bf16* __restrict__ dst, int n) {
    int i = (blockIdx.x * blockDim.x + threadIdx.x) * 4;
    if (i < n) {
        float4 f = *(const float4*)(src + i);
        v4bf o = { (__bf16)f.x, (__bf16)f.y, (__bf16)f.z, (__bf16)f.w };
        *(v4bf*)(dst + i) = o;
    }
}

// ---------------------------------------------------------------------------
// 2) Fused QKV projection + RoPE.
//    X[8192,1024]bf16 @ Wcat[3072,1024]^T bf16 -> Q,K rope'd to [B,H,S,64],
//    V transposed to [B,H,64,S].  One wave = 32x64 output tile (2 m-subtiles
//    share every B load -> half the weight-side L2 traffic).
// ---------------------------------------------------------------------------
__global__ __launch_bounds__(128) void qkv_rope_gemm(
    const __bf16* __restrict__ xb, const __bf16* __restrict__ wcat,
    const int* __restrict__ tok,
    __bf16* __restrict__ qb, __bf16* __restrict__ kb, __bf16* __restrict__ vtb) {

    const int wave = threadIdx.x >> 5;
    const int lane = threadIdx.x & 31;
    const int col  = lane & 15;
    const int half = lane >> 4;

    const int m0 = blockIdx.x * 128 + wave * 32;  // row-tile base (b*s flat)
    const int n0 = blockIdx.y * 64;               // col-tile base over 3*D

    const __bf16* Arow0 = xb + (size_t)(m0 + col) * D_;       // A row = lane&15
    const __bf16* Arow1 = xb + (size_t)(m0 + 16 + col) * D_;

    v8f acc[2][4] = {};
    for (int k = 0; k < D_; k += 32) {
        // prefetch A 8 k-tiles (256 elems) ahead toward WGP$ (global_prefetch_b8)
        __builtin_prefetch(Arow0 + k + 256, 0, 0);
        __builtin_prefetch(Arow1 + k + 256, 0, 0);
        v16bf a0 = mk16(*(const v8bf*)(Arow0 + k + half * 16),
                        *(const v8bf*)(Arow0 + k + half * 16 + 8));
        v16bf a1 = mk16(*(const v8bf*)(Arow1 + k + half * 16),
                        *(const v8bf*)(Arow1 + k + half * 16 + 8));
#pragma unroll
        for (int t = 0; t < 4; ++t) {
            v16bf b = *(const v16bf*)(wcat + (size_t)(n0 + t * 16 + col) * D_ + k + half * 16);
            acc[0][t] = wmma_bf16(a0, b, acc[0][t]);
            acc[1][t] = wmma_bf16(a1, b, acc[1][t]);
        }
    }

    const int which = n0 >> 10;        // 0=q, 1=k, 2=v
    const int nin   = n0 & 1023;
    const int h     = nin >> 6;
    const int b     = m0 >> 11;        // / 2048  (32-row tile never crosses b)

#pragma unroll
    for (int u = 0; u < 2; ++u) {
        const int s_base = (m0 + u * 16) & 2047;
        float posv[8];
#pragma unroll
        for (int r = 0; r < 8; ++r) posv[r] = (float)tok[s_base + r + 8 * half];

#pragma unroll
        for (int t = 0; t < 4; ++t) {
            const int dl = t * 16 + col;       // dim within head (0..63)
            if (which < 2) {
                // RoPE: pairs (2i, 2i+1) are adjacent lanes in the C layout.
                const float invf = __powf(10000.0f, -(float)(dl >> 1) * (1.0f / 32.0f));
                __bf16* dst = (which == 0) ? qb : kb;
#pragma unroll
                for (int r = 0; r < 8; ++r) {
                    float v = acc[u][t][r];
                    float p = __shfl_xor(v, 1, 32);      // partner of the pair
                    float sn, cs;
                    __sincosf(posv[r] * invf, &sn, &cs);
                    float o = v * cs + ((dl & 1) ? p * sn : -p * sn);
                    int s = s_base + r + 8 * half;
                    dst[((size_t)(b * H_ + h) * S_ + s) * DH_ + dl] = (__bf16)o;
                }
            } else {
#pragma unroll
                for (int r = 0; r < 8; ++r) {
                    int s = s_base + r + 8 * half;
                    vtb[((size_t)(b * H_ + h) * DH_ + dl) * S_ + s] = (__bf16)acc[u][t][r];
                }
            }
        }
    }
}

// ---------------------------------------------------------------------------
// 3) Flash attention: one wave = one (b,h) x 32 query rows (two 16-row
//    subtiles sharing the K operand registers).  Streams 32 keys per chunk:
//    QK^T via WMMA, online softmax in accumulator layout (16-lane shuffle
//    reductions), P transposed through per-wave LDS (explicit s_wait_dscnt),
//    then P@V via WMMA against V^T.
// ---------------------------------------------------------------------------
__global__ __launch_bounds__(128) void flash_attn(
    const __bf16* __restrict__ qb, const __bf16* __restrict__ kb,
    const __bf16* __restrict__ vtb, __bf16* __restrict__ attnb) {

    __shared__ __bf16 pbuf[4][16][32];          // 1 KB per wave

    const int wave = threadIdx.x >> 5;
    const int lane = threadIdx.x & 31;
    const int col  = lane & 15;
    const int half = lane >> 4;

    const int wgid = blockIdx.x * 4 + wave;
    const int qt   = wgid % (S_ / 32);
    const int bh   = wgid / (S_ / 32);
    const int q0   = qt * 32;

    const __bf16* Q  = qb  + (size_t)bh * S_ * DH_;
    const __bf16* K  = kb  + (size_t)bh * S_ * DH_;
    const __bf16* VT = vtb + (size_t)bh * DH_ * S_;
    __bf16 (*pw)[32] = pbuf[wave];

    // Q subtiles in A-operand layout (row = lane&15, K chunks per half).
    v16bf aq[2][2];
#pragma unroll
    for (int u = 0; u < 2; ++u) {
        const __bf16* qrow = Q + (size_t)(q0 + u * 16 + col) * DH_;
        aq[u][0] = mk16(*(const v8bf*)(qrow + half * 16),
                        *(const v8bf*)(qrow + half * 16 + 8));
        aq[u][1] = mk16(*(const v8bf*)(qrow + 32 + half * 16),
                        *(const v8bf*)(qrow + 40 + half * 16));
    }

    float Mrow[2][8], Lrow[2][8];
#pragma unroll
    for (int u = 0; u < 2; ++u)
#pragma unroll
        for (int r = 0; r < 8; ++r) { Mrow[u][r] = -3.0e38f; Lrow[u][r] = 0.0f; }
    v8f o[2][4] = {};

    for (int kv = 0; kv <= q0 + 31; kv += 32) {
        // ---- K operand: 32 keys x 64 dims, cached in regs for both subtiles ----
        const __bf16* kr0 = K + (size_t)(kv + col) * DH_;
        const __bf16* kr1 = K + (size_t)(kv + 16 + col) * DH_;
        __builtin_prefetch(kr0 + 32 * DH_, 0, 0);   // next chunk's K lines
        __builtin_prefetch(kr1 + 32 * DH_, 0, 0);
        const v16bf kb00 = *(const v16bf*)(kr0 + half * 16);
        const v16bf kb01 = *(const v16bf*)(kr0 + 32 + half * 16);
        const v16bf kb10 = *(const v16bf*)(kr1 + half * 16);
        const v16bf kb11 = *(const v16bf*)(kr1 + 32 + half * 16);

#pragma unroll
        for (int u = 0; u < 2; ++u) {
            v8f s0 = {}, s1 = {};
            s0 = wmma_bf16(aq[u][0], kb00, s0);
            s0 = wmma_bf16(aq[u][1], kb01, s0);
            s1 = wmma_bf16(aq[u][0], kb10, s1);
            s1 = wmma_bf16(aq[u][1], kb11, s1);

            // ---- scale + causal mask + online softmax ----
#pragma unroll
            for (int r = 0; r < 8; ++r) {
                const int row = q0 + u * 16 + r + 8 * half;
                float v0 = s0[r] * 0.125f;           // 1/sqrt(64)
                float v1 = s1[r] * 0.125f;
                v0 = (kv + col      > row) ? -1.0e9f : v0;
                v1 = (kv + 16 + col > row) ? -1.0e9f : v1;
                float mx = rmax16(fmaxf(v0, v1));
                float Mn = fmaxf(Mrow[u][r], mx);
                float alpha = __expf(Mrow[u][r] - Mn);
                float p0 = __expf(v0 - Mn);
                float p1 = __expf(v1 - Mn);
                Lrow[u][r] = Lrow[u][r] * alpha + rsum16(p0 + p1);
                Mrow[u][r] = Mn;
                o[u][0][r] *= alpha; o[u][1][r] *= alpha;
                o[u][2][r] *= alpha; o[u][3][r] *= alpha;
                pw[r + 8 * half][col]      = (__bf16)p0;   // C-layout -> LDS
                pw[r + 8 * half][col + 16] = (__bf16)p1;
            }
            // Wave-local LDS transpose: cross-lane stores visible before reload.
            // (DS ops are in-order per wave, so the later stores of the next
            //  subtile cannot bypass these loads -> only the RAW wait needed.)
            asm volatile("s_wait_dscnt 0" ::: "memory");

            // P reloaded in A-operand layout (row = lane&15, K chunk per half).
            v16bf ap = mk16(*(const v8bf*)&pw[col][half * 16],
                            *(const v8bf*)&pw[col][half * 16 + 8]);

            // ---- O += P @ V : 16x64 output, K = 32 keys ----
#pragma unroll
            for (int t = 0; t < 4; ++t) {
                v16bf bv = *(const v16bf*)(VT + (size_t)(t * 16 + col) * S_ + kv + half * 16);
                o[u][t] = wmma_bf16(ap, bv, o[u][t]);
            }
        }
    }

    // ---- normalize and scatter to [B, S, H*DH] (bf16) for the O-projection ----
    const int bb = bh >> 4, hh = bh & 15;
#pragma unroll
    for (int u = 0; u < 2; ++u)
#pragma unroll
        for (int t = 0; t < 4; ++t)
#pragma unroll
            for (int r = 0; r < 8; ++r) {
                const int s = q0 + u * 16 + r + 8 * half;
                float val = o[u][t][r] / Lrow[u][r];
                attnb[((size_t)bb * S_ + s) * D_ + hh * DH_ + t * 16 + col] = (__bf16)val;
            }
}

// ---------------------------------------------------------------------------
// 4) Output projection: attn[8192,1024]bf16 @ o_w[1024,1024]^T -> fp32 out.
//    Same 32x64 per-wave tiling as the QKV GEMM.
// ---------------------------------------------------------------------------
__global__ __launch_bounds__(128) void oproj_gemm(
    const __bf16* __restrict__ attnb, const __bf16* __restrict__ wob,
    float* __restrict__ out) {

    const int wave = threadIdx.x >> 5;
    const int lane = threadIdx.x & 31;
    const int col  = lane & 15;
    const int half = lane >> 4;

    const int m0 = blockIdx.x * 128 + wave * 32;
    const int n0 = blockIdx.y * 64;

    const __bf16* Arow0 = attnb + (size_t)(m0 + col) * D_;
    const __bf16* Arow1 = attnb + (size_t)(m0 + 16 + col) * D_;

    v8f acc[2][4] = {};
    for (int k = 0; k < D_; k += 32) {
        __builtin_prefetch(Arow0 + k + 256, 0, 0);
        __builtin_prefetch(Arow1 + k + 256, 0, 0);
        v16bf a0 = mk16(*(const v8bf*)(Arow0 + k + half * 16),
                        *(const v8bf*)(Arow0 + k + half * 16 + 8));
        v16bf a1 = mk16(*(const v8bf*)(Arow1 + k + half * 16),
                        *(const v8bf*)(Arow1 + k + half * 16 + 8));
#pragma unroll
        for (int t = 0; t < 4; ++t) {
            v16bf b = *(const v16bf*)(wob + (size_t)(n0 + t * 16 + col) * D_ + k + half * 16);
            acc[0][t] = wmma_bf16(a0, b, acc[0][t]);
            acc[1][t] = wmma_bf16(a1, b, acc[1][t]);
        }
    }
#pragma unroll
    for (int u = 0; u < 2; ++u)
#pragma unroll
        for (int t = 0; t < 4; ++t)
#pragma unroll
            for (int r = 0; r < 8; ++r) {
                out[(size_t)(m0 + u * 16 + r + 8 * half) * D_ + n0 + t * 16 + col] = acc[u][t][r];
            }
}

// ---------------------------------------------------------------------------
// Host launcher
// ---------------------------------------------------------------------------
extern "C" void kernel_launch(void* const* d_in, const int* in_sizes, int n_in,
                              void* d_out, int out_size, void* d_ws, size_t ws_size,
                              hipStream_t stream) {
    (void)in_sizes; (void)n_in; (void)out_size; (void)ws_size;

    const float* x   = (const float*)d_in[0];
    const int*   tok = (const int*)d_in[1];
    const float* qw  = (const float*)d_in[2];
    const float* kw  = (const float*)d_in[3];
    const float* vw  = (const float*)d_in[4];
    const float* ow  = (const float*)d_in[5];

    // Workspace carve-up (~92 MB total, 256B aligned slices).
    size_t off = 0;
    auto carve = [&](size_t bytes) -> void* {
        void* p = (char*)d_ws + off;
        off += (bytes + 255) & ~(size_t)255;
        return p;
    };
    __bf16* xb    = (__bf16*)carve((size_t)M_ * D_ * 2);          // x in bf16
    __bf16* wcat  = (__bf16*)carve((size_t)3 * D_ * D_ * 2);      // [q_w;k_w;v_w]
    __bf16* wob   = (__bf16*)carve((size_t)D_ * D_ * 2);          // o_w
    __bf16* qb    = (__bf16*)carve((size_t)M_ * D_ * 2);          // Q [B,H,S,64]
    __bf16* kb    = (__bf16*)carve((size_t)M_ * D_ * 2);          // K [B,H,S,64]
    __bf16* vtb   = (__bf16*)carve((size_t)M_ * D_ * 2);          // V^T [B,H,64,S]
    __bf16* attnb = (__bf16*)carve((size_t)M_ * D_ * 2);          // attn out bf16

    const int thr = 256;
    cvt_bf16_kernel<<<(M_ * D_ / 4 + thr - 1) / thr, thr, 0, stream>>>(x, xb, M_ * D_);
    cvt_bf16_kernel<<<(D_ * D_ / 4 + thr - 1) / thr, thr, 0, stream>>>(qw, wcat,               D_ * D_);
    cvt_bf16_kernel<<<(D_ * D_ / 4 + thr - 1) / thr, thr, 0, stream>>>(kw, wcat + D_ * D_,     D_ * D_);
    cvt_bf16_kernel<<<(D_ * D_ / 4 + thr - 1) / thr, thr, 0, stream>>>(vw, wcat + 2 * D_ * D_, D_ * D_);
    cvt_bf16_kernel<<<(D_ * D_ / 4 + thr - 1) / thr, thr, 0, stream>>>(ow, wob,                D_ * D_);

    qkv_rope_gemm<<<dim3(M_ / 128, (3 * D_) / 64), 128, 0, stream>>>(xb, wcat, tok, qb, kb, vtb);

    flash_attn<<<dim3((B_ * H_ * (S_ / 32)) / 4), 128, 0, stream>>>(qb, kb, vtb, attnb);

    oproj_gemm<<<dim3(M_ / 128, D_ / 64), 128, 0, stream>>>(attnb, wob, (float*)d_out);
}
